// CrossAttention_75883482186328
// MI455X (gfx1250) — compile-verified, matrix operand order checked by reading, not run on hardware
//
#include <hip/hip_runtime.h>
#include <hip/hip_bf16.h>

// ---------------------------------------------------------------------------
// CrossAttention: S = X X^T (CxC, K=N), column-softmax(S), out = A * dst
// B=8, C=512, N=64*64=4096.  bf16 WMMA (v_wmma_f32_16x16x32_bf16), f32 accum.
// 64x64 output tile per wave (4x4 of 16x16): 16 WMMA per 16 b128 loads.
// __launch_bounds__(32,1): one-wave blocks, full VGPR budget, no spills.
// ---------------------------------------------------------------------------

typedef __attribute__((ext_vector_type(16))) __bf16        v16bf;
typedef __attribute__((ext_vector_type(8)))  float         v8f;
typedef __attribute__((ext_vector_type(4)))  unsigned int  u32x4;
typedef __attribute__((ext_vector_type(4)))  float         f32x4;

constexpr int BATCH = 8;
constexpr int C     = 512;
constexpr int N     = 64 * 64;   // 4096

// ---- f32 -> bf16 round-to-nearest-even --------------------------------------
__device__ __forceinline__ unsigned short f2bf(float x) {
    unsigned int u = __float_as_uint(x);
    u += 0x7FFFu + ((u >> 16) & 1u);
    return (unsigned short)(u >> 16);
}

// ---- src: plain f32 -> bf16 convert (layout preserved) ----------------------
__global__ void cvt_f32_to_bf16_kernel(const float* __restrict__ in,
                                       unsigned short* __restrict__ out,
                                       long n) {
    long i = (long)blockIdx.x * blockDim.x + threadIdx.x;
    long stride = (long)gridDim.x * blockDim.x;
    for (; i < n; i += stride) out[i] = f2bf(in[i]);
}

// ---- dst: f32 [C][N] -> bf16 transposed [N][C], LDS 32x32 tile --------------
__global__ void transpose_cvt_kernel(const float* __restrict__ in,
                                     unsigned short* __restrict__ out) {
    __shared__ unsigned short tile[32][33];
    const int b  = blockIdx.z;
    const int n0 = blockIdx.x * 32;
    const int c0 = blockIdx.y * 32;
    const float* src = in + (long)b * C * N;
    unsigned short* dstT = out + (long)b * N * C;

    for (int r = threadIdx.y; r < 32; r += 8)
        tile[r][threadIdx.x] = f2bf(src[(long)(c0 + r) * N + n0 + threadIdx.x]);
    __syncthreads();
    for (int r = threadIdx.y; r < 32; r += 8)
        dstT[(long)(n0 + r) * C + c0 + threadIdx.x] = tile[threadIdx.x][r];
}

// ---- bf16 16x32 fragment loader (row-major source) --------------------------
// ISA 16-bit A-matrix 16x32 layout (CDNA5 ISA §7.12.2):
//   lane L: row m = L&15, half = L>>4
//   slots 0..7  -> k = half*8 + 0..7       (one contiguous 16B run)
//   slots 8..15 -> k = 16 + half*8 + 0..7  (one contiguous 16B run)
// => two global_load_b128 per lane. B (32x16, col n = L&15) is the same
//    pattern applied to the transposed operand.
__device__ __forceinline__ v16bf load_frag_rm(const unsigned short* __restrict__ base,
                                              int ldRow, int lane) {
    const int m    = lane & 15;
    const int half = lane >> 4;
    const unsigned short* q = base + (long)m * ldRow + half * 8;
    union { u32x4 x[2]; v16bf v; } f;
    f.x[0] = *(const u32x4*)(q);
    f.x[1] = *(const u32x4*)(q + 16);
    return f.v;
}

__device__ __forceinline__ v8f wmma_bf16(v16bf a, v16bf b, v8f c) {
    return __builtin_amdgcn_wmma_f32_16x16x32_bf16(
        /*neg_a=*/false, a, /*neg_b=*/false, b,
        /*c_mod=*/(short)0, c, /*reuse_a=*/false, /*reuse_b=*/false);
}

// C/D 16x16 f32 layout: lane L -> col = L&15 ; VGPR r -> row = r + (L>>4)*8
__device__ __forceinline__ void store_tile16x16(float* __restrict__ t, int ld,
                                                v8f acc, int lane) {
    const int col    = lane & 15;
    const int rowoff = (lane >> 4) * 8;
#pragma unroll
    for (int r = 0; r < 8; ++r) t[(long)(rowoff + r) * ld + col] = acc[r];
}

// ---- Kernel: Gram matrix S[b] = Xb * Xb^T  ----------------------------------
// one wave per 64x64 output tile: 4x4 of 16x16 WMMA tiles, K loop = 4096/32
__global__ void __launch_bounds__(32, 1)
gram_gemm_kernel(const unsigned short* __restrict__ srcb,
                 float* __restrict__ S) {
    const int lane = threadIdx.x;            // block = 1 wave
    const int tj   = blockIdx.x;             // 64-col tile
    const int ti   = blockIdx.y;             // 64-row tile
    const int b    = blockIdx.z;

    const unsigned short* X = srcb + (long)b * C * N;
    const unsigned short* Abase = X + (long)(ti * 64) * N;
    const unsigned short* Bbase = X + (long)(tj * 64) * N;  // col j of XX^T == row j of X

    v8f acc[4][4] = {};

    for (int k0 = 0; k0 < N; k0 += 32) {
        __builtin_prefetch(Abase + k0 + 32, 0, 3);
        __builtin_prefetch(Bbase + k0 + 32, 0, 3);
        v16bf bf[4];
#pragma unroll
        for (int n = 0; n < 4; ++n)
            bf[n] = load_frag_rm(Bbase + (long)(n * 16) * N + k0, N, lane);
#pragma unroll
        for (int m = 0; m < 4; ++m) {
            v16bf af = load_frag_rm(Abase + (long)(m * 16) * N + k0, N, lane);
#pragma unroll
            for (int n = 0; n < 4; ++n)
                acc[m][n] = wmma_bf16(af, bf[n], acc[m][n]);
        }
    }

    float* out = S + ((long)b * C + ti * 64) * C + tj * 64;
#pragma unroll
    for (int m = 0; m < 4; ++m)
#pragma unroll
        for (int n = 0; n < 4; ++n)
            store_tile16x16(out + (long)(m * 16) * C + n * 16, C, acc[m][n], lane);
}

// ---- Kernel: softmax over axis=1 (columns of S) -> bf16 A'[i][j] ------------
// S is symmetric, so column j == row j: read contiguous rows (f32x4 loads),
// write normalized column j (strided bf16 stores, L2-absorbed).
__global__ void __launch_bounds__(256)
col_softmax_kernel(const float* __restrict__ S,
                   unsigned short* __restrict__ Ap) {
    const int gtid = blockIdx.x * blockDim.x + threadIdx.x;
    const int wave = gtid >> 5;
    const int lane = gtid & 31;
    const int b    = wave / C;
    const int j    = wave % C;

    const float* row = S + ((long)b * C + j) * C;   // == column j by symmetry
    f32x4 v[4];
    float mx = -3.402823466e+38f;
#pragma unroll
    for (int r = 0; r < 4; ++r) {
        v[r] = *(const f32x4*)(row + lane * 4 + 128 * r);
#pragma unroll
        for (int e = 0; e < 4; ++e) mx = fmaxf(mx, v[r][e]);
    }
#pragma unroll
    for (int o = 16; o > 0; o >>= 1) mx = fmaxf(mx, __shfl_xor(mx, o, 32));

    float sum = 0.0f;
#pragma unroll
    for (int r = 0; r < 4; ++r)
#pragma unroll
        for (int e = 0; e < 4; ++e) { v[r][e] = __expf(v[r][e] - mx); sum += v[r][e]; }
#pragma unroll
    for (int o = 16; o > 0; o >>= 1) sum += __shfl_xor(sum, o, 32);

    const float inv = __frcp_rn(sum);
    unsigned short* out = Ap + (long)b * C * C + j;  // A'[i][j], row-major over i
#pragma unroll
    for (int r = 0; r < 4; ++r)
#pragma unroll
        for (int e = 0; e < 4; ++e)
            out[(long)(lane * 4 + 128 * r + e) * C] = f2bf(v[r][e] * inv);
}

// ---- Kernel: out[b] = A'[b] (CxC) * dst[b] (CxN), dst given transposed ------
// B-matrix element B[k][n] = dstT[n][k]: row n of dstT -> same row-major loader.
__global__ void __launch_bounds__(32, 1)
out_gemm_kernel(const unsigned short* __restrict__ Ap,
                const unsigned short* __restrict__ dstT,
                float* __restrict__ out) {
    const int lane = threadIdx.x;
    const int tn   = blockIdx.x;   // 64-wide col tile over N
    const int ti   = blockIdx.y;   // 64-high row tile over C
    const int b    = blockIdx.z;

    const unsigned short* Abase = Ap   + ((long)b * C + ti * 64) * C;
    const unsigned short* Bbase = dstT + ((long)b * N + tn * 64) * C;

    v8f acc[4][4] = {};

    for (int k0 = 0; k0 < C; k0 += 32) {
        __builtin_prefetch(Abase + k0 + 32, 0, 3);
        __builtin_prefetch(Bbase + k0 + 32, 0, 3);
        v16bf bf[4];
#pragma unroll
        for (int n = 0; n < 4; ++n)
            bf[n] = load_frag_rm(Bbase + (long)(n * 16) * C + k0, C, lane);
#pragma unroll
        for (int m = 0; m < 4; ++m) {
            v16bf af = load_frag_rm(Abase + (long)(m * 16) * C + k0, C, lane);
#pragma unroll
            for (int n = 0; n < 4; ++n)
                acc[m][n] = wmma_bf16(af, bf[n], acc[m][n]);
        }
    }

    float* o = out + ((long)b * C + ti * 64) * N + tn * 64;
#pragma unroll
    for (int m = 0; m < 4; ++m)
#pragma unroll
        for (int n = 0; n < 4; ++n)
            store_tile16x16(o + (long)(m * 16) * N + n * 16, N, acc[m][n], lane);
}

// ---------------------------------------------------------------------------
extern "C" void kernel_launch(void* const* d_in, const int* in_sizes, int n_in,
                              void* d_out, int out_size, void* d_ws, size_t ws_size,
                              hipStream_t stream) {
    const float* src = (const float*)d_in[0];
    const float* dst = (const float*)d_in[1];
    float* out = (float*)d_out;

    const long nElem = (long)BATCH * C * N;        // 16,777,216

    // workspace layout (76 MB total)
    unsigned short* srcb = (unsigned short*)d_ws;                      // 32 MB
    unsigned short* dstT = srcb + nElem;                               // 32 MB
    float*          S    = (float*)(dstT + nElem);                     //  8 MB
    unsigned short* Ap   = (unsigned short*)(S + (long)BATCH * C * C); //  4 MB

    // 1) convert src to bf16; convert+transpose dst to bf16 [N][C]
    cvt_f32_to_bf16_kernel<<<2048, 256, 0, stream>>>(src, srcb, nElem);
    transpose_cvt_kernel<<<dim3(N / 32, C / 32, BATCH), dim3(32, 8), 0, stream>>>(dst, dstT);

    // 2) Gram matrix per batch: one wave per 64x64 tile
    gram_gemm_kernel<<<dim3(C / 64, C / 64, BATCH), 32, 0, stream>>>(srcb, S);

    // 3) column softmax: one wave per (b, j)
    col_softmax_kernel<<<(BATCH * C) / 8, 256, 0, stream>>>(S, Ap);

    // 4) out = A' * dst : one wave per 64x64 tile
    out_gemm_kernel<<<dim3(N / 64, C / 64, BATCH), 32, 0, stream>>>(Ap, dstT, out);
}
